// ModulatedConv2d_19387482374500
// MI455X (gfx1250) — matrix operand order repaired and checked by reading, not compiled, not run on hardware
//
#include <hip/hip_runtime.h>
#include <hip/hip_bf16.h>

typedef __attribute__((ext_vector_type(16))) __bf16 v16bf;
typedef __attribute__((ext_vector_type(8)))  float  v8f;

#define BATCH 16
#define CIN   256
#define COUT  256
#define HH    64
#define WW    64
#define WD    256

__device__ __forceinline__ unsigned short f2bf(float f) {
    unsigned int u = __builtin_bit_cast(unsigned int, f);
    unsigned int r = u + 0x7FFFu + ((u >> 16) & 1u);
    return (unsigned short)(r >> 16);
}

// ---------------- prep kernel 1: s[b,c] = w[b,:] . style_w[c,:] + style_b[c]
__global__ void __launch_bounds__(256) style_kernel(
    const float* __restrict__ w, const float* __restrict__ style_w,
    const float* __restrict__ style_b, float* __restrict__ s)
{
    __shared__ float wv[WD];
    const int b = blockIdx.x;
    const int c = threadIdx.x;
    wv[c] = w[b * WD + c];
    __syncthreads();
    float acc = style_b[c];
    #pragma unroll 8
    for (int d = 0; d < WD; ++d) acc += wv[d] * style_w[c * WD + d];
    s[b * CIN + c] = acc;
}

// ---------------- prep kernel 2: wsq[cout,cin] = scale^2 * sum_kk cw^2 ; pack bf16 weights [tap][cout][cin]
__global__ void __launch_bounds__(256) wsqpack_kernel(
    const float* __restrict__ cw, const float* __restrict__ scale,
    float* __restrict__ wsq, unsigned short* __restrict__ wpk)
{
    const int id = blockIdx.x * 256 + threadIdx.x;   // id = cout*256 + cin
    const float sc = scale[0];
    float sum = 0.f;
    #pragma unroll
    for (int t = 0; t < 9; ++t) {
        float v = cw[(size_t)id * 9 + t];
        sum += v * v;
        wpk[(size_t)t * (COUT * CIN) + id] = f2bf(v * sc);
    }
    wsq[id] = sum * sc * sc;
}

// ---------------- prep kernel 3: demod[b,cout] = rsqrt( sum_cin s^2 * wsq + 1e-8 )
__global__ void __launch_bounds__(256) demod_kernel(
    const float* __restrict__ s, const float* __restrict__ wsq,
    float* __restrict__ demod)
{
    __shared__ float s2[CIN];
    const int b = blockIdx.x;
    const int n = threadIdx.x;
    float sv = s[b * CIN + n];
    s2[n] = sv * sv;
    __syncthreads();
    float acc = 1e-8f;
    #pragma unroll 8
    for (int c = 0; c < CIN; ++c) acc += s2[c] * wsq[n * CIN + c];
    demod[b * COUT + n] = rsqrtf(acc);
}

// ---------------- prep kernel 4: xb = bf16( s[b,cin] * x )   (elementwise, float4 vectorized)
__global__ void __launch_bounds__(256) modx_kernel(
    const float* __restrict__ x, const float* __restrict__ s,
    unsigned short* __restrict__ xb)
{
    const int id = blockIdx.x * 256 + threadIdx.x;
    const int e  = id * 4;                       // element index, 16M total
    const int b  = e >> 20;                      // / (256*4096)
    const int c  = (e >> 12) & 255;              // / 4096 % 256
    const float sv = s[b * CIN + c];
    float4 v = *(const float4*)(x + e);
    uint2 o;
    o.x = (unsigned int)f2bf(v.x * sv) | ((unsigned int)f2bf(v.y * sv) << 16);
    o.y = (unsigned int)f2bf(v.z * sv) | ((unsigned int)f2bf(v.w * sv) << 16);
    *(uint2*)(xb + e) = o;
}

// ---------------- main kernel: implicit-GEMM 3x3 conv via bf16 WMMA
// block: 256 threads (8 waves, 2(M) x 4(N) grid). Tile: 1 batch, 1 output row h
// (M = 64 pixels), N = 128 couts. Each wave: 32 pixels x 32 couts = 2x2 WMMA tiles
// (A and B fragments each reused twice -> 2 ds_load_b128 per WMMA).
// K reduction = 3(ky) x 8(cin chunks of 32) x 3(kx); each (chunk,tap) = one WMMA K=32 step.
__global__ void __launch_bounds__(256) conv_wmma_kernel(
    const unsigned short* __restrict__ xb,    // [16][256][64][64] bf16 (modulated)
    const unsigned short* __restrict__ wpk,   // [9][256][256] bf16 (scaled weights, [tap][cout][cin])
    const float* __restrict__ demod,          // [16][256]
    float* __restrict__ out)                  // [16][256][64][64]
{
    // A: [pixel -1..64][cin 0..31], row stride 40 (80B, 16B aligned, bank-spread)
    __shared__ unsigned short As[66][40];
    // B: [kx][cout 0..127][cin 0..31], same padded stride
    __shared__ unsigned short Bs[3][128][40];

    const int tid   = threadIdx.x;
    const int lane  = tid & 31;
    const int wid   = tid >> 5;
    const int waveM = wid & 1;           // pixel half (32 each)
    const int waveN = wid >> 1;          // cout quarter (32 each)
    const int khalf = lane >> 4;         // K-half selector per WMMA layout
    const int cb    = blockIdx.x * 128;  // cout base
    const int h     = blockIdx.y;
    const int bz    = blockIdx.z;

    // staging roles
    const int aC = tid >> 3;             // 0..31 cin row (A)
    const int aP = (tid & 7) * 8;        // pixel group (A)

    // permanent zero halo rows (w = -1 and w = 64)
    if (tid < 32)                 As[0][tid]       = 0;
    else if (tid < 64)            As[65][tid - 32] = 0;

    v8f acc[2][2] = {};

    for (int ky = 0; ky < 3; ++ky) {
        const int  hs  = h + ky - 1;
        const bool hOK = (hs >= 0) && (hs < HH);
        for (int cc = 0; cc < CIN; cc += 32) {
            // prefetch next chunk's source lines into cache while this chunk computes
            if (cc + 32 < CIN) {
                if (hOK)
                    __builtin_prefetch(xb + ((((size_t)bz * CIN + cc + 32 + aC) * HH + hs) * WW + aP), 0, 0);
                __builtin_prefetch(wpk + (((size_t)(ky * 3) * COUT + cb + (tid >> 1)) * CIN + cc + 32 + (tid & 1) * 16), 0, 0);
            }
            __syncthreads();   // prev iter's fragment reads done before restaging

            // ---- stage A tile: 64 pixels x 32 cin (transposed store -> pixel-major rows)
            {
                uint4 q = make_uint4(0u, 0u, 0u, 0u);
                if (hOK)
                    q = *(const uint4*)(xb + ((((size_t)bz * CIN + cc + aC) * HH + hs) * WW + aP));
                const unsigned short* e = (const unsigned short*)&q;
                #pragma unroll
                for (int j = 0; j < 8; ++j)
                    As[aP + 1 + j][aC] = e[j];
            }
            // ---- stage B tiles for the 3 kx taps: 128 cout x 32 cin each
            #pragma unroll
            for (int kx = 0; kx < 3; ++kx) {
                const int tap = ky * 3 + kx;
                #pragma unroll
                for (int ssub = 0; ssub < 2; ++ssub) {
                    const int slot = tid + ssub * 256;   // 512 uint4 slots
                    const int row  = slot >> 2;          // 0..127 cout row
                    const int g    = (slot & 3) * 8;     // cin group
                    uint4 q = *(const uint4*)(wpk + (((size_t)tap * COUT + cb + row) * CIN + cc + g));
                    *(uint4*)&Bs[kx][row][g] = q;
                }
            }
            __syncthreads();

            // ---- 3 WMMA K-steps (one per kx), A reused with +kx row shift
            #pragma unroll
            for (int kx = 0; kx < 3; ++kx) {
                union Frag { v16bf v; uint4 q[2]; };
                Frag fa[2], fb[2];
                #pragma unroll
                for (int mi = 0; mi < 2; ++mi) {
                    const unsigned short* arow = &As[waveM * 32 + mi * 16 + (lane & 15) + kx][0];
                    fa[mi].q[0] = *(const uint4*)(arow + khalf * 8);        // K {0-7 | 8-15}
                    fa[mi].q[1] = *(const uint4*)(arow + khalf * 8 + 16);   // K {16-23 | 24-31}
                }
                #pragma unroll
                for (int ni = 0; ni < 2; ++ni) {
                    const unsigned short* brow = &Bs[kx][waveN * 32 + ni * 16 + (lane & 15)][0];
                    fb[ni].q[0] = *(const uint4*)(brow + khalf * 8);        // K {0-7 | 8-15}
                    fb[ni].q[1] = *(const uint4*)(brow + khalf * 8 + 16);   // K {16-23 | 24-31}
                }
                #pragma unroll
                for (int mi = 0; mi < 2; ++mi)
                    #pragma unroll
                    for (int ni = 0; ni < 2; ++ni)
                        acc[mi][ni] = __builtin_amdgcn_wmma_f32_16x16x32_bf16(
                            false, fa[mi].v, false, fb[ni].v, (short)0, acc[mi][ni], false, false);
            }
        }
    }

    // ---- epilogue: apply demod per cout, scatter to NCHW
    #pragma unroll
    for (int mi = 0; mi < 2; ++mi) {
        const int wbase = waveM * 32 + mi * 16 + khalf * 8;
        #pragma unroll
        for (int ni = 0; ni < 2; ++ni) {
            const int n = cb + waveN * 32 + ni * 16 + (lane & 15);
            const float dm = demod[bz * COUT + n];
            float* o = out + (((size_t)bz * COUT + n) * HH + h) * WW + wbase;
            #pragma unroll
            for (int i = 0; i < 8; ++i)
                o[i] = acc[mi][ni][i] * dm;
        }
    }
}

extern "C" void kernel_launch(void* const* d_in, const int* in_sizes, int n_in,
                              void* d_out, int out_size, void* d_ws, size_t ws_size,
                              hipStream_t stream) {
    const float* x        = (const float*)d_in[0];   // [16,256,64,64]
    const float* w        = (const float*)d_in[1];   // [16,256]
    const float* cw       = (const float*)d_in[2];   // [256,256,3,3]
    const float* style_w  = (const float*)d_in[3];   // [256,256]
    const float* style_b  = (const float*)d_in[4];   // [256]
    const float* scale    = (const float*)d_in[5];   // [1]
    float* out            = (float*)d_out;

    // workspace layout
    char* ws = (char*)d_ws;
    float* s_buf     = (float*)(ws);                                  // 4096 f32
    float* demod_buf = (float*)(ws + 16384);                          // 4096 f32
    float* wsq_buf   = (float*)(ws + 32768);                          // 65536 f32
    unsigned short* wpk = (unsigned short*)(ws + 32768 + 262144);     // 9*256*256 bf16
    unsigned short* xb  = (unsigned short*)(ws + 32768 + 262144 + 1179648); // 16M bf16

    style_kernel  <<<BATCH, 256, 0, stream>>>(w, style_w, style_b, s_buf);
    wsqpack_kernel<<<(COUT * CIN) / 256, 256, 0, stream>>>(cw, scale, wsq_buf, wpk);
    demod_kernel  <<<BATCH, 256, 0, stream>>>(s_buf, wsq_buf, demod_buf);
    modx_kernel   <<<(BATCH * CIN * HH * WW) / (4 * 256), 256, 0, stream>>>(x, s_buf, xb);

    dim3 grid(COUT / 128, HH, BATCH);
    conv_wmma_kernel<<<grid, 256, 0, stream>>>(xb, wpk, demod_buf, out);
}
